// SwinTransformerBlock_MS_14800457302061
// MI455X (gfx1250) — matrix operand order repaired and checked by reading, not compiled, not run on hardware
//
#include <hip/hip_runtime.h>

#define DIM   96
#define HSZ   256
#define WSZ   256
#define BSZ   2
#define WS    8
#define SS    4
#define NHEAD 6
#define PHD   16
#define NTOK  131072   // BSZ*HSZ*WSZ

typedef _Float16 f16;
typedef __attribute__((ext_vector_type(16))) _Float16 v16h;
typedef __attribute__((ext_vector_type(8)))  _Float16 v8h;
typedef __attribute__((ext_vector_type(8)))  float    v8f;

static __device__ __forceinline__ v8f zero_v8f() {
  v8f z;
#pragma unroll
  for (int i = 0; i < 8; ++i) z[i] = 0.0f;
  return z;
}

static __device__ __forceinline__ v8h zero_v8h() {
  v8h z;
#pragma unroll
  for (int i = 0; i < 8; ++i) z[i] = (f16)0.0f;
  return z;
}

static __device__ __forceinline__ v16h combine(v8h lo, v8h hi) {
  v16h r;
#pragma unroll
  for (int i = 0; i < 8; ++i) { r[i] = lo[i]; r[8 + i] = hi[i]; }
  return r;
}

static __device__ __forceinline__ v8h cvt8(float4 a, float4 b) {
  v8h r;
  r[0] = (f16)a.x; r[1] = (f16)a.y; r[2] = (f16)a.z; r[3] = (f16)a.w;
  r[4] = (f16)b.x; r[5] = (f16)b.y; r[6] = (f16)b.z; r[7] = (f16)b.w;
  return r;
}

static __device__ __forceinline__ v8h cvt8s(float4 a, float4 b, float s) {
  v8h r;
  r[0] = (f16)(a.x * s); r[1] = (f16)(a.y * s); r[2] = (f16)(a.z * s); r[3] = (f16)(a.w * s);
  r[4] = (f16)(b.x * s); r[5] = (f16)(b.y * s); r[6] = (f16)(b.z * s); r[7] = (f16)(b.w * s);
  return r;
}

// A fragment (16x32, M x K) from a row-major tile with leading dim ld halfs.
// Contiguous k order (k = 16*(lane/16) + s) -> each lane reads two contiguous 16B chunks (ds_load_b128).
static __device__ __forceinline__ v16h load_a_frag(const f16* base, int ld, int lane) {
  const int m  = lane & 15;
  const int kb = (lane < 16) ? 0 : 16;
  const f16* row = base + m * ld + kb;
  return combine(*(const v8h*)row, *(const v8h*)(row + 8));
}

// B fragment stored in "fragment layout": 16 contiguous halfs per lane.
static __device__ __forceinline__ v16h load_frag_contig(const f16* p) {
  return combine(*(const v8h*)p, *(const v8h*)(p + 8));
}

// B fragment with stride between K elements (scattered): b[k] = p[k*stride]
static __device__ __forceinline__ v16h load_b_strided(const f16* p, int stride) {
  v16h b;
#pragma unroll
  for (int k = 0; k < 16; ++k) b[k] = p[k * stride];
  return b;
}

// ---------------------------------------------------------------------------
// LayerNorm over C=96 per token (float4 vectorized)
// ---------------------------------------------------------------------------
__global__ void layernorm_kernel(const float* __restrict__ x, const float* __restrict__ g,
                                 const float* __restrict__ b, float* __restrict__ y, int tokens) {
  int t = blockIdx.x * blockDim.x + threadIdx.x;
  if (t >= tokens) return;
  const float4* p4 = (const float4*)(x + (size_t)t * DIM);
  float4 v[24];
#pragma unroll
  for (int c = 0; c < 24; ++c) v[c] = p4[c];
  float m = 0.f;
#pragma unroll
  for (int c = 0; c < 24; ++c) m += v[c].x + v[c].y + v[c].z + v[c].w;
  m *= (1.0f / DIM);
  float var = 0.f;
#pragma unroll
  for (int c = 0; c < 24; ++c) {
    float dx = v[c].x - m, dy = v[c].y - m, dz = v[c].z - m, dw = v[c].w - m;
    var += dx * dx + dy * dy + dz * dz + dw * dw;
  }
  var *= (1.0f / DIM);
  float inv = rsqrtf(var + 1e-5f);
  const float4* g4 = (const float4*)g;
  const float4* b4 = (const float4*)b;
  float4* o4 = (float4*)(y + (size_t)t * DIM);
#pragma unroll
  for (int c = 0; c < 24; ++c) {
    float4 gg = g4[c], bb = b4[c], o;
    o.x = (v[c].x - m) * inv * gg.x + bb.x;
    o.y = (v[c].y - m) * inv * gg.y + bb.y;
    o.z = (v[c].z - m) * inv * gg.z + bb.z;
    o.w = (v[c].w - m) * inv * gg.w + bb.w;
    o4[c] = o;
  }
}

// ---------------------------------------------------------------------------
// Direct NHWC conv (small channel counts), weights OIHW, optional lrelu(0.2)
// ---------------------------------------------------------------------------
__global__ void conv2d_nhwc_kernel(const float* __restrict__ in, const float* __restrict__ w,
                                   const float* __restrict__ bias, float* __restrict__ out,
                                   int Cin, int Cout, int ks, int pad, int lrelu) {
  int idx = blockIdx.x * blockDim.x + threadIdx.x;
  int total = BSZ * HSZ * WSZ * Cout;
  if (idx >= total) return;
  int co = idx % Cout;
  int t  = idx / Cout;
  int x  = t % WSZ; t /= WSZ;
  int y  = t % HSZ;
  int b  = t / HSZ;
  const int kss = ks * ks;
  float s = bias[co];
  for (int kh = 0; kh < ks; ++kh) {
    int yy = y + kh - pad;
    if ((unsigned)yy >= (unsigned)HSZ) continue;
    for (int kw = 0; kw < ks; ++kw) {
      int xx = x + kw - pad;
      if ((unsigned)xx >= (unsigned)WSZ) continue;
      const float4* ip4 = (const float4*)(in + (((size_t)b * HSZ + yy) * WSZ + xx) * Cin);
      const float* wp = w + (size_t)co * Cin * kss + kh * ks + kw;
      for (int c4 = 0; c4 < (Cin >> 2); ++c4) {
        float4 u = ip4[c4];
        const float* wq = wp + (size_t)c4 * 4 * kss;
        s += u.x * wq[0] + u.y * wq[kss] + u.z * wq[2 * kss] + u.w * wq[3 * kss];
      }
    }
  }
  if (lrelu && s < 0.f) s *= 0.2f;
  out[(size_t)idx] = s;
}

// ---------------------------------------------------------------------------
// WMMA GEMM: out[M,N] = A[M,K] * W[N,K]^T + bias (+GELU) (+residual).
// Block 128 (4 waves), tile 64x64, k-step 32. B kept in LDS fragment layout.
// All 4 B fragments are loaded before the WMMA burst (one dscnt wait).
// ---------------------------------------------------------------------------
__global__ void __launch_bounds__(128) gemm_bias_act(
    const float* __restrict__ Af32, const f16* __restrict__ Af16,
    const float* __restrict__ Wt, const float* __restrict__ bias,
    const float* __restrict__ resid, float* __restrict__ outF, f16* __restrict__ outH,
    int M, int N, int K, int act) {
  __shared__ __align__(16) f16 Al[64 * 32];       // row-major A tile
  __shared__ __align__(16) f16 Bl[4 * 32 * 16];   // fragment-layout B tile
  const int tid  = threadIdx.x;
  const int lane = tid & 31;
  const int wv   = tid >> 5;
  const int m0   = blockIdx.x * 64;
  const int n0   = blockIdx.y * 64;

  // cooperative-load geometry
  const int ar = tid >> 1;           // A row 0..63
  const int ah = (tid & 1) * 16;     // A column half: 0 or 16
  const int bn  = lane & 15;
  const int bkb = (lane < 16) ? 0 : 16;
  const int bcol = n0 + wv * 16 + bn;          // writer thread's W row
  const bool bok = bcol < N;

  v8f acc[4];
#pragma unroll
  for (int i = 0; i < 4; ++i) acc[i] = zero_v8f();

  const int nk = K >> 5;
  for (int kc = 0; kc < nk; ++kc) {
    __syncthreads();
    {   // A tile: 16 consecutive floats (or halfs) per thread
      size_t gi = (size_t)(m0 + ar) * K + kc * 32 + ah;
      v8h lo, hi;
      if (Af32) {
        const float4* p = (const float4*)(Af32 + gi);
        lo = cvt8(p[0], p[1]); hi = cvt8(p[2], p[3]);
      } else {
        const v8h* p = (const v8h*)(Af16 + gi);
        lo = p[0]; hi = p[1];
      }
      *(v8h*)(Al + ar * 32 + ah)     = lo;
      *(v8h*)(Al + ar * 32 + ah + 8) = hi;
    }
    {   // B tile in fragment layout: thread == consuming (ni,lane) slot
      v8h lo, hi;
      if (bok) {
        const float4* p = (const float4*)(Wt + (size_t)bcol * K + kc * 32 + bkb);
        lo = cvt8(p[0], p[1]); hi = cvt8(p[2], p[3]);
      } else {
        lo = zero_v8h(); hi = zero_v8h();
      }
      *(v8h*)(Bl + tid * 16)     = lo;
      *(v8h*)(Bl + tid * 16 + 8) = hi;
    }
    __syncthreads();
    v16h af = load_a_frag(Al + wv * 16 * 32, 32, lane);
    v16h bf[4];
#pragma unroll
    for (int ni = 0; ni < 4; ++ni)
      bf[ni] = load_frag_contig(Bl + (ni * 32 + lane) * 16);
#pragma unroll
    for (int ni = 0; ni < 4; ++ni)
      acc[ni] = __builtin_amdgcn_wmma_f32_16x16x32_f16(false, af, false, bf[ni],
                                                       (short)0, acc[ni], false, false);
  }

  const int rlo = (lane < 16) ? 0 : 8;
  const int cl  = lane & 15;
#pragma unroll
  for (int ni = 0; ni < 4; ++ni) {
    int col = n0 + ni * 16 + cl;
    if (col >= N) continue;
#pragma unroll
    for (int r = 0; r < 8; ++r) {
      size_t row = (size_t)(m0 + wv * 16 + r + rlo);
      float v = acc[ni][r] + bias[col];
      if (act == 1) v = 0.5f * v * (1.0f + erff(v * 0.7071067811865475f));
      if (resid) v += resid[row * (size_t)N + col];
      if (outF) outF[row * (size_t)N + col] = v;
      if (outH) outH[row * (size_t)N + col] = (f16)v;
    }
  }
}

// ---------------------------------------------------------------------------
// Shifted-window attention. Block = 96 threads (3 waves = 3 heads),
// grid = 2048 windows * 2 head-groups. One wave = one (window, head).
// Q,K tiles zero-padded to K=32 in LDS so all fragments are ds_load_b128.
// ---------------------------------------------------------------------------
__global__ void __launch_bounds__(96) swin_attn(
    const float* __restrict__ qk1, const float* __restrict__ qk3,
    const float* __restrict__ qk5, const float* __restrict__ vbuf,
    const float* __restrict__ rpb_table, float* __restrict__ attn_out) {
  extern __shared__ char smem_raw[];
  f16* smem = (f16*)smem_raw;

  const int tid  = threadIdx.x;
  const int lane = tid & 31;
  const int h    = (blockIdx.x & 1) * 3 + (tid >> 5);   // head 0..5
  const int blk  = blockIdx.x >> 1;                     // window 0..2047
  const int bb   = blk >> 10;
  const int widx = blk & 1023;
  const int wh   = widx >> 5;
  const int ww   = widx & 31;

  // per-head LDS: Q 64x32 | K 64x32 | V 64x16 | P 64x64  = 9216 halfs (18KB)
  f16* Ql = smem + (tid >> 5) * 9216;
  f16* Kl = Ql + 2048;
  f16* Vl = Kl + 2048;
  f16* Pl = Vl + 1024;

  const int sc = h >> 1, hh = h & 1;
  const float* qkbuf = (sc == 0) ? qk1 : (sc == 1 ? qk3 : qk5);
  const int qbase = hh * 16;
  const int kbase = 32 + hh * 16;
  const int vbase = h * 16;

  const v8h z8 = zero_v8h();
  // gather q/k/v for this window+head with the roll(-SS) shift
#pragma unroll
  for (int tt = 0; tt < 2; ++tt) {
    int t = lane * 2 + tt;
    int i = t >> 3, j = t & 7;
    int y = (wh * 8 + i + SS) & 255;
    int x = (ww * 8 + j + SS) & 255;
    size_t pix = ((size_t)bb * HSZ + y) * WSZ + x;
    const float4* qp = (const float4*)(qkbuf + pix * 64 + qbase);
    const float4* kp = (const float4*)(qkbuf + pix * 64 + kbase);
    const float4* vp = (const float4*)(vbuf + pix * 96 + vbase);
    *(v8h*)(Ql + t * 32)      = cvt8s(qp[0], qp[1], 0.25f);   // PHD^-0.5
    *(v8h*)(Ql + t * 32 + 8)  = cvt8s(qp[2], qp[3], 0.25f);
    *(v8h*)(Ql + t * 32 + 16) = z8;
    *(v8h*)(Ql + t * 32 + 24) = z8;
    *(v8h*)(Kl + t * 32)      = cvt8(kp[0], kp[1]);
    *(v8h*)(Kl + t * 32 + 8)  = cvt8(kp[2], kp[3]);
    *(v8h*)(Kl + t * 32 + 16) = z8;
    *(v8h*)(Kl + t * 32 + 24) = z8;
    *(v8h*)(Vl + t * 16)      = cvt8(vp[0], vp[1]);
    *(v8h*)(Vl + t * 16 + 8)  = cvt8(vp[2], vp[3]);
  }
  __syncthreads();

  const int rlo = (lane < 16) ? 0 : 8;
  const int cl  = lane & 15;
  const int kbB = (lane < 16) ? 0 : 16;

  // Hoisted K^T fragments: B[kk][nn] = K[nn][kk]; row-contiguous after padding.
  v16h bfK[4];
#pragma unroll
  for (int ni = 0; ni < 4; ++ni)
    bfK[ni] = load_frag_contig(Kl + (ni * 16 + cl) * 32 + kbB);
  // Hoisted V fragments: B[kk][nn] = V[kk][nn] (strided gather, done once).
  v16h bfV[2];
#pragma unroll
  for (int kc = 0; kc < 2; ++kc)
    bfV[kc] = load_b_strided(Vl + (kc * 32 + kbB) * 16 + cl, 16);

  // S = Q K^T, bias+mask, row softmax, P -> LDS (f16)
#pragma unroll 1
  for (int mi = 0; mi < 4; ++mi) {
    v8f acc[4];
#pragma unroll
    for (int i = 0; i < 4; ++i) acc[i] = zero_v8f();
    v16h af = load_a_frag(Ql + mi * 16 * 32, 32, lane);
#pragma unroll
    for (int ni = 0; ni < 4; ++ni)
      acc[ni] = __builtin_amdgcn_wmma_f32_16x16x32_f16(false, af, false, bfK[ni],
                                                       (short)0, acc[ni], false, false);
#pragma unroll
    for (int r = 0; r < 8; ++r) {
      int n = mi * 16 + r + rlo;
      int i1 = n >> 3, j1 = n & 7;
      int y1 = wh * 8 + i1, x1 = ww * 8 + j1;
      int lab1 = ((y1 < 248) ? 0 : (y1 < 252 ? 1 : 2)) * 3 +
                 ((x1 < 248) ? 0 : (x1 < 252 ? 1 : 2));
      float s[4];
#pragma unroll
      for (int ni = 0; ni < 4; ++ni) {
        int m = ni * 16 + cl;
        int i2 = m >> 3, j2 = m & 7;
        int y2 = wh * 8 + i2, x2 = ww * 8 + j2;
        int lab2 = ((y2 < 248) ? 0 : (y2 < 252 ? 1 : 2)) * 3 +
                   ((x2 < 248) ? 0 : (x2 < 252 ? 1 : 2));
        float rpb = rpb_table[(size_t)((i1 - i2 + 7) * 15 + (j1 - j2 + 7)) * NHEAD + h];
        s[ni] = acc[ni][r] + rpb + ((lab1 == lab2) ? 0.0f : -100.0f);
      }
      float mx = fmaxf(fmaxf(s[0], s[1]), fmaxf(s[2], s[3]));
      mx = fmaxf(mx, __shfl_xor(mx, 1, 32));
      mx = fmaxf(mx, __shfl_xor(mx, 2, 32));
      mx = fmaxf(mx, __shfl_xor(mx, 4, 32));
      mx = fmaxf(mx, __shfl_xor(mx, 8, 32));
      float e[4], sum = 0.f;
#pragma unroll
      for (int ni = 0; ni < 4; ++ni) { e[ni] = __expf(s[ni] - mx); sum += e[ni]; }
      sum += __shfl_xor(sum, 1, 32);
      sum += __shfl_xor(sum, 2, 32);
      sum += __shfl_xor(sum, 4, 32);
      sum += __shfl_xor(sum, 8, 32);
      float inv = 1.0f / sum;
#pragma unroll
      for (int ni = 0; ni < 4; ++ni)
        Pl[n * 64 + ni * 16 + cl] = (f16)(e[ni] * inv);
    }
  }

  // O = P V ; scatter to final (unshifted) positions
#pragma unroll 1
  for (int mi = 0; mi < 4; ++mi) {
    v8f o = zero_v8f();
#pragma unroll
    for (int kc = 0; kc < 2; ++kc) {
      v16h af = load_a_frag(Pl + mi * 16 * 64 + kc * 32, 64, lane);
      o = __builtin_amdgcn_wmma_f32_16x16x32_f16(false, af, false, bfV[kc],
                                                 (short)0, o, false, false);
    }
#pragma unroll
    for (int r = 0; r < 8; ++r) {
      int n = mi * 16 + r + rlo;
      int i = n >> 3, j = n & 7;
      int y = (wh * 8 + i + SS) & 255;
      int x = (ww * 8 + j + SS) & 255;
      size_t pix = ((size_t)bb * HSZ + y) * WSZ + x;
      attn_out[pix * 96 + vbase + cl] = o[r];
    }
  }
}

// ---------------------------------------------------------------------------
// Launch. Workspace layout (256 MB):
//   [0,50.3M)       xn (LN1 out; later xn2)
//   [50.3M,83.9M)   qk1 out   } fc1's f16 output (100.66 MB) overlays these
//   [83.9M,117.4M)  qk3 out   } three exactly (dead after attention)
//   [117.4M,151M)   qk5 out   }
//   [151M,201.3M)   v out
//   [201.3M,209.7M) conv tmp1, [209.7M,218.1M) conv tmp2
//   [218.1M,268.4M) attention out (final positions)
//   x2 lives in d_out (fc2 reads its residual from d_out).
// ---------------------------------------------------------------------------
extern "C" void kernel_launch(void* const* d_in, const int* in_sizes, int n_in,
                              void* d_out, int out_size, void* d_ws, size_t ws_size,
                              hipStream_t stream) {
  (void)in_sizes; (void)n_in; (void)out_size; (void)ws_size;
  const float* qk1_w = (const float*)d_in[0];
  const float* qk1_b = (const float*)d_in[1];
  const float* qk3_w1 = (const float*)d_in[2];
  const float* qk3_b1 = (const float*)d_in[3];
  const float* qk3_w2 = (const float*)d_in[4];
  const float* qk3_b2 = (const float*)d_in[5];
  const float* qk3_w3 = (const float*)d_in[6];
  const float* qk3_b3 = (const float*)d_in[7];
  const float* qk5_w1 = (const float*)d_in[8];
  const float* qk5_b1 = (const float*)d_in[9];
  const float* qk5_w2 = (const float*)d_in[10];
  const float* qk5_b2 = (const float*)d_in[11];
  const float* qk5_w3 = (const float*)d_in[12];
  const float* qk5_b3 = (const float*)d_in[13];
  const float* v_w    = (const float*)d_in[14];
  const float* v_b    = (const float*)d_in[15];
  const float* rpb    = (const float*)d_in[16];
  const float* proj_w = (const float*)d_in[17];
  const float* proj_b = (const float*)d_in[18];
  const float* fc1_w  = (const float*)d_in[19];
  const float* fc1_b  = (const float*)d_in[20];
  const float* fc2_w  = (const float*)d_in[21];
  const float* fc2_b  = (const float*)d_in[22];
  const float* xin    = (const float*)d_in[23];
  const float* n1g    = (const float*)d_in[24];
  const float* n1b    = (const float*)d_in[25];
  const float* n2g    = (const float*)d_in[26];
  const float* n2b    = (const float*)d_in[27];

  char* ws = (char*)d_ws;
  float* xn   = (float*)(ws + 0);
  float* qk1o = (float*)(ws + 50331648UL);
  float* qk3o = (float*)(ws + 83886080UL);
  float* qk5o = (float*)(ws + 117440512UL);
  float* vo   = (float*)(ws + 150994944UL);
  float* t1   = (float*)(ws + 201326592UL);
  float* t2   = (float*)(ws + 209715200UL);
  float* attn = (float*)(ws + 218103808UL);
  f16*   hbuf = (f16*)(ws + 50331648UL);     // overlays qk1/qk3/qk5
  float* out  = (float*)d_out;

  const int M = NTOK;

  // 1) LN1
  layernorm_kernel<<<512, 256, 0, stream>>>(xin, n1g, n1b, xn, M);

  // 2) 1x1 convs as WMMA GEMMs: qk1 (96->64), v (96->96)
  gemm_bias_act<<<dim3(2048, 1), 128, 0, stream>>>(xn, nullptr, qk1_w, qk1_b,
      nullptr, qk1o, nullptr, M, 64, 96, 0);
  gemm_bias_act<<<dim3(2048, 2), 128, 0, stream>>>(xn, nullptr, v_w, v_b,
      nullptr, vo, nullptr, M, 96, 96, 0);

  // 3) 3x3 chain: 96->16 lrelu, 16->16 lrelu, 16->64
  conv2d_nhwc_kernel<<<8192, 256, 0, stream>>>(xn, qk3_w1, qk3_b1, t1, 96, 16, 3, 1, 1);
  conv2d_nhwc_kernel<<<8192, 256, 0, stream>>>(t1, qk3_w2, qk3_b2, t2, 16, 16, 1, 0, 1);
  conv2d_nhwc_kernel<<<32768, 256, 0, stream>>>(t2, qk3_w3, qk3_b3, qk3o, 16, 64, 3, 1, 0);

  // 4) 5x5 chain: 96->8 lrelu, 8->8 lrelu, 8->64
  conv2d_nhwc_kernel<<<4096, 256, 0, stream>>>(xn, qk5_w1, qk5_b1, t1, 96, 8, 5, 2, 1);
  conv2d_nhwc_kernel<<<4096, 256, 0, stream>>>(t1, qk5_w2, qk5_b2, t2, 8, 8, 1, 0, 1);
  conv2d_nhwc_kernel<<<32768, 256, 0, stream>>>(t2, qk5_w3, qk5_b3, qk5o, 8, 64, 5, 2, 0);

  // 5) shifted-window attention (WMMA), 3 heads/block, 54 KB LDS
  swin_attn<<<4096, 96, 3 * 9216 * sizeof(f16), stream>>>(qk1o, qk3o, qk5o, vo, rpb, attn);

  // 6) proj + residual(x) -> d_out (== x2)
  gemm_bias_act<<<dim3(2048, 2), 128, 0, stream>>>(attn, nullptr, proj_w, proj_b,
      xin, out, nullptr, M, 96, 96, 0);

  // 7) LN2 (x2 -> xn2)
  layernorm_kernel<<<512, 256, 0, stream>>>(out, n2g, n2b, xn, M);

  // 8) fc1 + exact GELU -> f16 h
  gemm_bias_act<<<dim3(2048, 6), 128, 0, stream>>>(xn, nullptr, fc1_w, fc1_b,
      nullptr, nullptr, hbuf, M, 384, 96, 1);

  // 9) fc2 + residual(x2) -> d_out
  gemm_bias_act<<<dim3(2048, 2), 128, 0, stream>>>(nullptr, hbuf, fc2_w, fc2_b,
      out, out, nullptr, M, 96, 384, 0);
}
